// TransformerBlock_79517024518650
// MI455X (gfx1250) — compile-verified, matrix operand order checked by reading, not run on hardware
//
#include <hip/hip_runtime.h>
#include <hip/hip_bf16.h>

// ---------------------------------------------------------------------------
// Problem constants (match reference)
// ---------------------------------------------------------------------------
constexpr int BB = 4;
constexpr int SS = 2048;
constexpr int DD = 1024;
constexpr int HH = 16;
constexpr int FF = 4096;
constexpr int DKK = 64;          // DD / HH

typedef __bf16 bf16;
typedef __attribute__((ext_vector_type(16))) bf16           v16bf;
typedef __attribute__((ext_vector_type(8)))  float          v8f;
typedef __attribute__((ext_vector_type(8)))  unsigned short ushort8;

union FragU { v16bf v; ushort8 h[2]; unsigned short u[16]; };

__device__ __forceinline__ unsigned short f2bf(float f) {
    union { float f; unsigned u; } x; x.f = f;
    unsigned r = x.u + 0x7FFFu + ((x.u >> 16) & 1u);   // round-to-nearest-even
    return (unsigned short)(r >> 16);
}
__device__ __forceinline__ unsigned pack2bf(float a, float b) {
    return (unsigned)f2bf(a) | ((unsigned)f2bf(b) << 16);
}

// Fragment whose 16 elements are K-contiguous in LDS/memory at `row` of a
// tile with `stride` (bf16 units, multiple of 8 => 16 B).  Per ISA 7.12.2:
// element e<8 -> K = half*8+e (16 contiguous bytes), e>=8 -> K = 16+half*8+e-8.
__device__ __forceinline__ v16bf load_frag_k(const unsigned short* base,
                                             int stride, int row, int half) {
    FragU f;
    const unsigned short* p = base + row * stride + half * 8;
    f.h[0] = *(const ushort8*)(p);          // two ds_load_b128
    f.h[1] = *(const ushort8*)(p + 16);
    return f.v;
}

__device__ __forceinline__ v8f wmma_bf16(v16bf a, v16bf b, v8f c) {
    return __builtin_amdgcn_wmma_f32_16x16x32_bf16(
        /*neg_a=*/false, a, /*neg_b=*/false, b,
        /*c_mod=*/(short)0, c, /*reuse_a=*/false, /*reuse_b=*/false);
}

// ---------------------------------------------------------------------------
// One-time fp32 -> bf16 convert (row-major, vectorized)
// ---------------------------------------------------------------------------
__global__ __launch_bounds__(256)
void f32_to_bf16_kernel(const float* __restrict__ src, unsigned short* __restrict__ dst)
{
    size_t i = (size_t)blockIdx.x * 256 + threadIdx.x;
    float4 s = *(const float4*)&src[i * 4];
    uint2 p; p.x = pack2bf(s.x, s.y); p.y = pack2bf(s.z, s.w);
    *(uint2*)&dst[i * 4] = p;
}

// ---------------------------------------------------------------------------
// One-time fp32 [R,C] -> bf16 transposed [C,R] (tiled through LDS)
// ---------------------------------------------------------------------------
__global__ __launch_bounds__(256)
void transpose_to_bf16_kernel(const float* __restrict__ src, unsigned short* __restrict__ dst,
                              int R, int C)
{
    __shared__ unsigned short t[64 * 72];
    const int tid = threadIdx.x;
    const int r0 = blockIdx.y * 64, c0 = blockIdx.x * 64;
#pragma unroll
    for (int i = 0; i < 4; ++i) {
        int e = tid + i * 256;                // 1024 float4 chunks
        int r = e >> 4, c = (e & 15) * 4;
        float4 s4 = *(const float4*)&src[(size_t)(r0 + r) * C + c0 + c];
        uint2 p; p.x = pack2bf(s4.x, s4.y); p.y = pack2bf(s4.z, s4.w);
        *(uint2*)&t[r * 72 + c] = p;
    }
    __syncthreads();
#pragma unroll
    for (int i = 0; i < 4; ++i) {
        int e = tid + i * 256;                // output [C,R] in 4-wide chunks
        int r = e >> 4, c = (e & 15) * 4;     // r = out row (src col), c = out col (src row)
        uint2 p;
        p.x = (unsigned)t[(c + 0) * 72 + r] | ((unsigned)t[(c + 1) * 72 + r] << 16);
        p.y = (unsigned)t[(c + 2) * 72 + r] | ((unsigned)t[(c + 3) * 72 + r] << 16);
        *(uint2*)&dst[(size_t)(c0 + r) * R + r0 + c] = p;
    }
}

// ---------------------------------------------------------------------------
// Tiled GEMM:  C[M,N] = A[M,K] * W[K,N] + bias, A and Wt in bf16, Wt is the
// PRE-TRANSPOSED weight [N,K].  256 threads = 8 waves, tile 128x128x32; each
// wave owns a 32x64 sub-tile (2x4 WMMA tiles).  Both LDS tiles are [row][k]
// with stride 40 so every fragment is two ds_load_b128.
// MODE: 0 = bias -> bf16 out          1 = bias+relu -> bf16 out
//       2 = bias+resid(f32) -> f32    3 = bias -> bf16 V-transposed [B,H,DK,S]
// ---------------------------------------------------------------------------
constexpr int TM = 128, TN = 128, TK = 32;
constexpr int TSTR = 40;    // bf16 units; 80 B pitch (16 B multiple)

template <int MODE>
__global__ __launch_bounds__(256)
void gemm_bf16_kernel(const unsigned short* __restrict__ A,
                      const unsigned short* __restrict__ Wt,
                      const float* __restrict__ bias, const float* __restrict__ resid,
                      float* __restrict__ Cf, unsigned short* __restrict__ Cb,
                      int M, int N, int K)
{
    __shared__ unsigned short Alds[TM * TSTR];   // [m][k]
    __shared__ unsigned short Btld[TN * TSTR];   // [n][k]

    const int tid  = threadIdx.x;
    const int wave = tid >> 5;
    const int lane = tid & 31;
    const int half = (lane >> 4) & 1;
    const int ln   = lane & 15;

    const int wm = wave >> 1;      // 0..3  -> 32 rows each
    const int wn = wave & 1;       // 0..1  -> 64 cols each

    const int row0 = blockIdx.y * TM;
    const int col0 = blockIdx.x * TN;

    v8f zero = {};
    v8f acc[2][4];
#pragma unroll
    for (int mi = 0; mi < 2; ++mi)
#pragma unroll
        for (int ni = 0; ni < 4; ++ni) acc[mi][ni] = zero;

    for (int k0 = 0; k0 < K; k0 += TK) {
        __syncthreads();
        // 512 ushort8 chunks per tile, 2 per thread: pure copy, no conversion
#pragma unroll
        for (int i = 0; i < 2; ++i) {
            int e = tid + i * 256;
            int r = e >> 2, c = (e & 3) * 8;
            *(ushort8*)&Alds[r * TSTR + c] =
                *(const ushort8*)&A[(size_t)(row0 + r) * K + k0 + c];
            *(ushort8*)&Btld[r * TSTR + c] =
                *(const ushort8*)&Wt[(size_t)(col0 + r) * K + k0 + c];
        }
        if (k0 + TK < K) {   // hint next tiles into cache (global_prefetch_b8)
            __builtin_prefetch(&A[(size_t)(row0 + (tid >> 1)) * K + k0 + TK], 0, 0);
            __builtin_prefetch(&Wt[(size_t)(col0 + (tid >> 1)) * K + k0 + TK], 0, 0);
        }
        __syncthreads();

        v16bf af[2];
#pragma unroll
        for (int mi = 0; mi < 2; ++mi)
            af[mi] = load_frag_k(Alds, TSTR, wm * 32 + mi * 16 + ln, half);
#pragma unroll
        for (int ni = 0; ni < 4; ++ni) {
            v16bf bfr = load_frag_k(Btld, TSTR, wn * 64 + ni * 16 + ln, half);
#pragma unroll
            for (int mi = 0; mi < 2; ++mi)
                acc[mi][ni] = wmma_bf16(af[mi], bfr, acc[mi][ni]);
        }
    }

    // ---- epilogue ----
#pragma unroll
    for (int mi = 0; mi < 2; ++mi) {
#pragma unroll
        for (int ni = 0; ni < 4; ++ni) {
#pragma unroll
            for (int r = 0; r < 8; ++r) {
                int grow = row0 + wm * 32 + mi * 16 + r + 8 * half;
                int gcol = col0 + wn * 64 + ni * 16 + ln;
                float v = acc[mi][ni][r] + bias[gcol];
                if (MODE == 0) {
                    Cb[(size_t)grow * N + gcol] = f2bf(v);
                } else if (MODE == 1) {
                    Cb[(size_t)grow * N + gcol] = f2bf(fmaxf(v, 0.0f));
                } else if (MODE == 2) {
                    Cf[(size_t)grow * N + gcol] = v + resid[(size_t)grow * N + gcol];
                } else {                 // V projection -> [B,H,DK,S] bf16
                    int bb = grow >> 11;          // / SS
                    int s  = grow & (SS - 1);
                    int hh = gcol >> 6;
                    int dk = gcol & 63;
                    Cb[((size_t)(bb * HH + hh) * DKK + dk) * SS + s] = f2bf(v);
                }
            }
        }
    }
}

// ---------------------------------------------------------------------------
// Causal flash attention.  One block per (b, h, 64-query-row tile).
// 128 threads = 4 waves; each wave owns a 16-row query strip.
// q/k: bf16 [B,S,D] (head h at col h*64).  v: bf16 [B,H,DK,S] (pre-transposed
// by the V-projection GEMM).  Output: bf16 [B,S,D].
// K tile [kv][dk] and V tile [dk][kv] both stride 72 -> b128 staging + frags.
// ---------------------------------------------------------------------------
constexpr int KSTR = 72;

__global__ __launch_bounds__(128)
void attn_kernel(const unsigned short* __restrict__ qg,
                 const unsigned short* __restrict__ kg,
                 const unsigned short* __restrict__ vt,
                 unsigned short* __restrict__ og)
{
    __shared__ unsigned short Kt [64 * KSTR];    // [kv][dk]
    __shared__ unsigned short Vtt[64 * KSTR];    // [dk][kv]
    __shared__ unsigned short Pb[4][16 * KSTR];  // per-wave P strip [m][kv]

    const int qt  = blockIdx.x;           // query tile (0..31)
    const int bh  = blockIdx.y;           // 0..B*H-1
    const int b   = bh / HH;
    const int h   = bh % HH;
    const int tid = threadIdx.x;
    const int wave = tid >> 5;
    const int lane = tid & 31;
    const int half = (lane >> 4) & 1;
    const int ln   = lane & 15;

    const int qbase = qt * 64;
    const int qrow0 = qbase + wave * 16;
    const float scale = 0.125f;           // 1/sqrt(64), applied to raw scores

    // ---- Q fragments straight from global bf16 (b128 loads) ----
    v16bf aq[2];
    {
        const unsigned short* qp =
            qg + (size_t)(b * SS + qrow0 + ln) * DD + h * DKK;
#pragma unroll
        for (int t = 0; t < 2; ++t) {
            FragU f;
            f.h[0] = *(const ushort8*)(qp + t * 32 + half * 8);
            f.h[1] = *(const ushort8*)(qp + t * 32 + half * 8 + 16);
            aq[t] = f.v;
        }
    }

    v8f zero = {};
    v8f oacc[4];
#pragma unroll
    for (int n = 0; n < 4; ++n) oacc[n] = zero;
    float mi[8], li[8];
#pragma unroll
    for (int r = 0; r < 8; ++r) { mi[r] = -1e30f; li[r] = 0.0f; }

    for (int jt = 0; jt <= qt; ++jt) {
        const int kb = jt * 64;
        __syncthreads();
        // ---- stage K [kv][dk] and V [dk][kv] tiles: pure b128 copies ----
#pragma unroll
        for (int i = 0; i < 4; ++i) {
            int e = tid + i * 128;                 // 512 ushort8 chunks
            int r = e >> 3, c = (e & 7) * 8;
            *(ushort8*)&Kt[r * KSTR + c] =
                *(const ushort8*)&kg[(size_t)(b * SS + kb + r) * DD + h * DKK + c];
            *(ushort8*)&Vtt[r * KSTR + c] =
                *(const ushort8*)&vt[((size_t)(bh * DKK) + r) * SS + kb + c];
        }
        __syncthreads();

        // ---- raw scores S = Q K^T : 16x64 per wave ----
        v8f sc[4];
#pragma unroll
        for (int n = 0; n < 4; ++n) sc[n] = zero;
#pragma unroll
        for (int t = 0; t < 2; ++t) {
#pragma unroll
            for (int n = 0; n < 4; ++n) {
                v16bf bk = load_frag_k(Kt + t * 32, KSTR, n * 16 + ln, half);
                sc[n] = wmma_bf16(aq[t], bk, sc[n]);
            }
        }
        // ---- scale + causal mask on the diagonal tile ----
#pragma unroll
        for (int n = 0; n < 4; ++n)
#pragma unroll
            for (int r = 0; r < 8; ++r) sc[n][r] *= scale;
        if (jt == qt) {
#pragma unroll
            for (int n = 0; n < 4; ++n)
#pragma unroll
                for (int r = 0; r < 8; ++r) {
                    int qr = qrow0 + r + 8 * half;
                    int kc = kb + n * 16 + ln;
                    if (kc > qr) sc[n][r] = -1e30f;
                }
        }
        // ---- online softmax, per row (row lives in 16 lanes of a half) ----
        unsigned short* pb = Pb[wave];
#pragma unroll
        for (int r = 0; r < 8; ++r) {
            float rmax = -1e30f;
#pragma unroll
            for (int n = 0; n < 4; ++n) rmax = fmaxf(rmax, sc[n][r]);
            rmax = fmaxf(rmax, __shfl_xor(rmax, 8));
            rmax = fmaxf(rmax, __shfl_xor(rmax, 4));
            rmax = fmaxf(rmax, __shfl_xor(rmax, 2));
            rmax = fmaxf(rmax, __shfl_xor(rmax, 1));
            float mnew  = fmaxf(mi[r], rmax);
            float alpha = __expf(mi[r] - mnew);
            float rsum  = 0.0f;
#pragma unroll
            for (int n = 0; n < 4; ++n) {
                float p = __expf(sc[n][r] - mnew);
                sc[n][r] = p;
                rsum += p;
            }
            rsum += __shfl_xor(rsum, 8);
            rsum += __shfl_xor(rsum, 4);
            rsum += __shfl_xor(rsum, 2);
            rsum += __shfl_xor(rsum, 1);
            li[r] = li[r] * alpha + rsum;
            mi[r] = mnew;
#pragma unroll
            for (int n = 0; n < 4; ++n) oacc[n][r] *= alpha;
            // spill P strip to wave-private LDS (C-layout -> A-layout relay)
#pragma unroll
            for (int n = 0; n < 4; ++n)
                pb[(r + 8 * half) * KSTR + n * 16 + ln] = f2bf(sc[n][r]);
        }
        // wave-internal LDS RAW: split dependency counter wait (CDNA5)
        asm volatile("s_wait_dscnt 0" ::: "memory");

        // ---- O += P * V ----
        v16bf ap[2];
#pragma unroll
        for (int t = 0; t < 2; ++t)
            ap[t] = load_frag_k(pb + t * 32, KSTR, ln, half);
#pragma unroll
        for (int t = 0; t < 2; ++t) {
#pragma unroll
            for (int n = 0; n < 4; ++n) {
                v16bf bv = load_frag_k(Vtt + t * 32, KSTR, n * 16 + ln, half);
                oacc[n] = wmma_bf16(ap[t], bv, oacc[n]);
            }
        }
    }

    // ---- write O / l as bf16 [B,S,D] ----
#pragma unroll
    for (int n = 0; n < 4; ++n)
#pragma unroll
        for (int r = 0; r < 8; ++r) {
            int grow = qrow0 + r + 8 * half;
            og[(size_t)(b * SS + grow) * DD + h * DKK + n * 16 + ln] =
                f2bf(oacc[n][r] / li[r]);
        }
}

// ---------------------------------------------------------------------------
// Row-wise LayerNorm over D=1024, one 256-thread block per row (float4 I/O).
// Optionally emits a bf16 copy for the next GEMM's A operand.
// ---------------------------------------------------------------------------
__global__ __launch_bounds__(256)
void ln_kernel(const float* __restrict__ x, const float* __restrict__ g,
               const float* __restrict__ b, float* __restrict__ out,
               unsigned short* __restrict__ out_bf)
{
    __shared__ float sred[8], ssred[8];
    const int row  = blockIdx.x;
    const int tid  = threadIdx.x;
    const int wave = tid >> 5;
    const int lane = tid & 31;
    const float* xr = x + (size_t)row * DD;

    float4 v4 = *(const float4*)&xr[tid * 4];
    float s  = v4.x + v4.y + v4.z + v4.w;
    float ss = v4.x * v4.x + v4.y * v4.y + v4.z * v4.z + v4.w * v4.w;
#pragma unroll
    for (int off = 16; off >= 1; off >>= 1) {
        s  += __shfl_xor(s, off);
        ss += __shfl_xor(ss, off);
    }
    if (lane == 0) { sred[wave] = s; ssred[wave] = ss; }
    __syncthreads();
    float ts = 0.0f, tss = 0.0f;
#pragma unroll
    for (int w = 0; w < 8; ++w) { ts += sred[w]; tss += ssred[w]; }
    float mu   = ts * (1.0f / DD);
    float var  = tss * (1.0f / DD) - mu * mu;
    float rstd = rsqrtf(var + 1e-5f);

    float4 g4 = *(const float4*)&g[tid * 4];
    float4 b4 = *(const float4*)&b[tid * 4];
    float4 o4;
    o4.x = (v4.x - mu) * rstd * g4.x + b4.x;
    o4.y = (v4.y - mu) * rstd * g4.y + b4.y;
    o4.z = (v4.z - mu) * rstd * g4.z + b4.z;
    o4.w = (v4.w - mu) * rstd * g4.w + b4.w;
    *(float4*)&out[(size_t)row * DD + tid * 4] = o4;
    if (out_bf) {
        uint2 p; p.x = pack2bf(o4.x, o4.y); p.y = pack2bf(o4.z, o4.w);
        *(uint2*)&out_bf[(size_t)row * DD + tid * 4] = p;
    }
}

// ---------------------------------------------------------------------------
// Host-side launch
// ---------------------------------------------------------------------------
extern "C" void kernel_launch(void* const* d_in, const int* in_sizes, int n_in,
                              void* d_out, int out_size, void* d_ws, size_t ws_size,
                              hipStream_t stream)
{
    (void)in_sizes; (void)n_in; (void)out_size; (void)ws_size;
    const float* x  = (const float*)d_in[0];
    /* d_in[1] = causal mask, implicit in the attention kernel */
    const float* Wq = (const float*)d_in[2];
    const float* bq = (const float*)d_in[3];
    const float* Wk = (const float*)d_in[4];
    const float* bk = (const float*)d_in[5];
    const float* Wv = (const float*)d_in[6];
    const float* bv = (const float*)d_in[7];
    const float* Wo = (const float*)d_in[8];
    const float* bo = (const float*)d_in[9];
    const float* g1 = (const float*)d_in[10];
    const float* b1 = (const float*)d_in[11];
    const float* W1 = (const float*)d_in[12];
    const float* c1 = (const float*)d_in[13];
    const float* W2 = (const float*)d_in[14];
    const float* c2 = (const float*)d_in[15];
    const float* g2 = (const float*)d_in[16];
    const float* b2 = (const float*)d_in[17];
    float* out = (float*)d_out;

    char* ws = (char*)d_ws;
    const size_t MB = 1ull << 20;
    typedef unsigned short u16;
    u16*   xb    = (u16*)  (ws +   0 * MB);   // 16 MB  bf16 x
    u16*   Wqt   = (u16*)  (ws +  16 * MB);   //  2 MB  bf16 [N,K]
    u16*   Wkt   = (u16*)  (ws +  18 * MB);
    u16*   Wvt   = (u16*)  (ws +  20 * MB);
    u16*   Wot   = (u16*)  (ws +  22 * MB);
    u16*   W1t   = (u16*)  (ws +  24 * MB);   //  8 MB
    u16*   W2t   = (u16*)  (ws +  32 * MB);   //  8 MB
    u16*   qb    = (u16*)  (ws +  40 * MB);   // 16 MB
    u16*   kb    = (u16*)  (ws +  56 * MB);   // 16 MB
    u16*   vtb   = (u16*)  (ws +  72 * MB);   // 16 MB  [B,H,DK,S]
    u16*   attnb = (u16*)  (ws +  88 * MB);   // 16 MB
    float* y     = (float*)(ws + 104 * MB);   // 32 MB  (reused for y2)
    float* x1    = (float*)(ws + 136 * MB);   // 32 MB
    u16*   x1b   = (u16*)  (ws + 168 * MB);   // 16 MB
    u16*   hbufb = (u16*)  (ws +  40 * MB);   // 64 MB, reuses q/k/v/attn
    float* y2    = (float*)(ws + 104 * MB);   // reuses y

    const int M = BB * SS;                    // 8192
    dim3 blk(256);

    // ---- one-time precision / layout pass ----
    f32_to_bf16_kernel<<<dim3((M * DD) / 1024), blk, 0, stream>>>(x, xb);
    transpose_to_bf16_kernel<<<dim3(DD / 64, DD / 64), blk, 0, stream>>>(Wq, Wqt, DD, DD);
    transpose_to_bf16_kernel<<<dim3(DD / 64, DD / 64), blk, 0, stream>>>(Wk, Wkt, DD, DD);
    transpose_to_bf16_kernel<<<dim3(DD / 64, DD / 64), blk, 0, stream>>>(Wv, Wvt, DD, DD);
    transpose_to_bf16_kernel<<<dim3(DD / 64, DD / 64), blk, 0, stream>>>(Wo, Wot, DD, DD);
    transpose_to_bf16_kernel<<<dim3(FF / 64, DD / 64), blk, 0, stream>>>(W1, W1t, DD, FF);
    transpose_to_bf16_kernel<<<dim3(DD / 64, FF / 64), blk, 0, stream>>>(W2, W2t, FF, DD);

    // ---- QKV projections (q,k row-major bf16; v per-head transposed) ----
    gemm_bf16_kernel<0><<<dim3(DD / TN, M / TM), blk, 0, stream>>>(xb, Wqt, bq, nullptr, nullptr, qb,  M, DD, DD);
    gemm_bf16_kernel<0><<<dim3(DD / TN, M / TM), blk, 0, stream>>>(xb, Wkt, bk, nullptr, nullptr, kb,  M, DD, DD);
    gemm_bf16_kernel<3><<<dim3(DD / TN, M / TM), blk, 0, stream>>>(xb, Wvt, bv, nullptr, nullptr, vtb, M, DD, DD);
    // ---- causal flash attention ----
    attn_kernel<<<dim3(SS / 64, BB * HH), dim3(128), 0, stream>>>(qb, kb, vtb, attnb);
    // ---- output projection + residual, layernorm 1 (dual f32/bf16 out) ----
    gemm_bf16_kernel<2><<<dim3(DD / TN, M / TM), blk, 0, stream>>>(attnb, Wot, bo, x, y, nullptr, M, DD, DD);
    ln_kernel<<<dim3(M), blk, 0, stream>>>(y, g1, b1, x1, x1b);
    // ---- FFN ----
    gemm_bf16_kernel<1><<<dim3(FF / TN, M / TM), blk, 0, stream>>>(x1b, W1t, c1, nullptr, nullptr, hbufb, M, FF, DD);
    gemm_bf16_kernel<2><<<dim3(DD / TN, M / TM), blk, 0, stream>>>(hbufb, W2t, c2, x1, y2, nullptr, M, DD, FF);
    ln_kernel<<<dim3(M), blk, 0, stream>>>(y2, g2, b2, out, nullptr);
}
